// FMRIROIConnectivityTransformer_14628658610334
// MI455X (gfx1250) — matrix operand order repaired and checked by reading, not compiled, not run on hardware
//
#include <hip/hip_runtime.h>

// ---------------------------------------------------------------------------
// MI455X (gfx1250): all GEMMs via v_wmma_f32_16x16x32_bf16, wave32.
// Each wave computes a 32x64 macro-tile (2 M-tiles x 4 N-tiles, 8 WMMAs per
// 32-K step) so A-fragments are reused 4x and B-fragments 2x -> ~21 FLOP/B
// of cache traffic, compute-bound against the bf16 WMMA ceiling.
// ---------------------------------------------------------------------------

#define DEV __device__ __forceinline__

typedef __attribute__((ext_vector_type(16))) __bf16 v16bf;
typedef __attribute__((ext_vector_type(8)))  float  v8f;

union FragA { v16bf v; uint4 u[2]; unsigned short s[16]; };
union FragC { v8f v; float f[8]; };

DEV unsigned short f2bf(float f) {           // round-to-nearest-even f32->bf16
  unsigned int u = __float_as_uint(f);
  u += 0x7FFFu + ((u >> 16) & 1u);
  return (unsigned short)(u >> 16);
}
DEV float bf2f(unsigned short h) { return __uint_as_float(((unsigned int)h) << 16); }
DEV float gelu_exact(float x) { return 0.5f * x * (1.0f + erff(x * 0.70710678118654752f)); }
DEV int imin(int a, int b) { return a < b ? a : b; }

// ---------------------------------------------------------------------------
// Generic WMMA GEMM:  C[M,N] = A[M,Kp] (bf16) * B^T[N,Kp] (bf16) + bias
// Optional: residual add, exact GELU, f32 store, bf16 store, transposed bf16
// store (to produce V^T with padded S for the attention kernel).
// One wave computes one 32x64 macro-tile. blockDim = 128 (4 waves).
// Requires M % 32 == 0, N % 64 == 0.  grid = (ceil(macroTiles/4), batch)
// ---------------------------------------------------------------------------
template<bool BIAS, bool RES, bool GEL, bool WF, bool WB, bool TST>
__global__ void __launch_bounds__(128)
gemm_bf16_wmma(const unsigned short* __restrict__ A, long long aBatch, int lda,
               const unsigned short* __restrict__ BT, long long bBatch, int ldb,
               const float* __restrict__ bias, int biasBatch,
               const float* __restrict__ resid, long long rBatch, int ldr,
               float* __restrict__ Cf, long long cfBatch, long long ldcf,
               unsigned short* __restrict__ Cb, long long cbBatch, int ldcb,
               int M, int N, int Kp,
               int tRows, int tLd, long long tBatch)
{
  int lane = threadIdx.x & 31;
  int wave = threadIdx.x >> 5;
  int mtiles = M >> 5, ntiles = N >> 6;          // 32x64 macro-tiles
  int tile = blockIdx.x * (blockDim.x >> 5) + wave;
  if (tile >= mtiles * ntiles) return;
  int mt = tile / ntiles, nt = tile - mt * ntiles;
  int m0 = mt << 5, n0 = nt << 6;
  int batch = blockIdx.y;
  int r = lane & 15, half = lane >> 4;

  const unsigned short* aRow0 = A + aBatch * batch + (long long)(m0 + r) * lda;
  const unsigned short* aRow1 = aRow0 + 16LL * lda;
  const unsigned short* bBase = BT + bBatch * batch;
  const unsigned short* bR0 = bBase + (long long)(n0 + r) * ldb;
  const unsigned short* bR1 = bR0 + 16LL * ldb;
  const unsigned short* bR2 = bR0 + 32LL * ldb;
  const unsigned short* bR3 = bR0 + 48LL * ldb;

  FragC c[2][4];
#pragma unroll
  for (int mi = 0; mi < 2; mi++)
#pragma unroll
    for (int j = 0; j < 4; j++)
#pragma unroll
      for (int i = 0; i < 8; i++) c[mi][j].f[i] = 0.f;

  for (int k0 = 0; k0 < Kp; k0 += 32) {
    FragA a0, a1;
    // A-fragment: rows m; K run [k0+16h, +15] contiguous
    a0.u[0] = *(const uint4*)(aRow0 + k0 + 16 * half);
    a0.u[1] = *(const uint4*)(aRow0 + k0 + 16 * half + 8);
    a1.u[0] = *(const uint4*)(aRow1 + k0 + 16 * half);
    a1.u[1] = *(const uint4*)(aRow1 + k0 + 16 * half + 8);
    // B-fragments: col n; K run [k0+16h, +15] contiguous in B^T
    FragA b0, b1, b2, b3;
    {
      const uint4* p0 = (const uint4*)(bR0 + k0 + 16 * half);
      const uint4* p1 = (const uint4*)(bR1 + k0 + 16 * half);
      const uint4* p2 = (const uint4*)(bR2 + k0 + 16 * half);
      const uint4* p3 = (const uint4*)(bR3 + k0 + 16 * half);
      b0.u[0] = p0[0]; b0.u[1] = p0[1];
      b1.u[0] = p1[0]; b1.u[1] = p1[1];
      b2.u[0] = p2[0]; b2.u[1] = p2[1];
      b3.u[0] = p3[0]; b3.u[1] = p3[1];
    }
    c[0][0].v = __builtin_amdgcn_wmma_f32_16x16x32_bf16(false, a0.v, false, b0.v, (short)0, c[0][0].v, false, false);
    c[1][0].v = __builtin_amdgcn_wmma_f32_16x16x32_bf16(false, a1.v, false, b0.v, (short)0, c[1][0].v, false, false);
    c[0][1].v = __builtin_amdgcn_wmma_f32_16x16x32_bf16(false, a0.v, false, b1.v, (short)0, c[0][1].v, false, false);
    c[1][1].v = __builtin_amdgcn_wmma_f32_16x16x32_bf16(false, a1.v, false, b1.v, (short)0, c[1][1].v, false, false);
    c[0][2].v = __builtin_amdgcn_wmma_f32_16x16x32_bf16(false, a0.v, false, b2.v, (short)0, c[0][2].v, false, false);
    c[1][2].v = __builtin_amdgcn_wmma_f32_16x16x32_bf16(false, a1.v, false, b2.v, (short)0, c[1][2].v, false, false);
    c[0][3].v = __builtin_amdgcn_wmma_f32_16x16x32_bf16(false, a0.v, false, b3.v, (short)0, c[0][3].v, false, false);
    c[1][3].v = __builtin_amdgcn_wmma_f32_16x16x32_bf16(false, a1.v, false, b3.v, (short)0, c[1][3].v, false, false);
  }

#pragma unroll
  for (int j = 0; j < 4; j++) {
    int col = n0 + j * 16 + r;
    float bv = 0.f;
    if (BIAS) bv = bias[(long long)biasBatch * batch + col];
#pragma unroll
    for (int mi = 0; mi < 2; mi++) {
#pragma unroll
      for (int i = 0; i < 8; i++) {
        int row = m0 + mi * 16 + i + 8 * half;
        float v = c[mi][j].f[i] + bv;
        if (RES) v += resid[rBatch * batch + (long long)row * ldr + col];
        if (GEL) v = gelu_exact(v);
        if (WF) Cf[cfBatch * batch + (long long)row * ldcf + col] = v;
        if (WB) {
          unsigned short hb = f2bf(v);
          if (TST) {   // store transposed per-sample: dst[b2][col][s]
            int b2 = row / tRows, s2 = row - b2 * tRows;
            Cb[cbBatch * batch + (long long)b2 * tBatch + (long long)col * tLd + s2] = hb;
          } else {
            Cb[cbBatch * batch + (long long)row * ldcb + col] = hb;
          }
        }
      }
    }
  }
}

// ---------------------------------------------------------------------------
// Fused self-attention for one (b, head, 16-query tile).  S=200 (pad 224),
// Dh=32.  Scores in LDS (f32), in-wave softmax, then P*V with WMMA.
// blockDim = 32 (one wave). grid = (13, H=6, B=256)
// ---------------------------------------------------------------------------
__global__ void __launch_bounds__(32)
attn_fused(const unsigned short* __restrict__ q,
           const unsigned short* __restrict__ k,
           const unsigned short* __restrict__ vT,   // [b][d=192][s pad 224]
           unsigned short* __restrict__ ao)          // [b][s][192]
{
  __shared__ float sc[16][224];
  const int S = 200, D = 192, SP = 224;
  int lane = threadIdx.x;
  int r = lane & 15, half = lane >> 4;
  int mt = blockIdx.x, hh = blockIdx.y, b = blockIdx.z;
  int m0 = mt * 16;
  long long base = (long long)b * S * D + hh * 32;

  // Q fragment (K=32 = one WMMA step); clamp padded query rows
  FragA a;
  {
    int qm = imin(m0 + r, S - 1);
    const unsigned short* qrow = q + base + (long long)qm * D;
    a.u[0] = *(const uint4*)(qrow + 16 * half);
    a.u[1] = *(const uint4*)(qrow + 16 * half + 8);
  }
  const float scale = 0.17677669529663687f;  // 1/sqrt(32)

  for (int nt = 0; nt < 13; nt++) {
    int ns = imin(nt * 16 + r, S - 1);      // clamp padded key rows
    const unsigned short* krow = k + base + (long long)ns * D + 16 * half;
    FragA bb;
    bb.u[0] = ((const uint4*)krow)[0];
    bb.u[1] = ((const uint4*)krow)[1];
    FragC c;
#pragma unroll
    for (int i = 0; i < 8; i++) c.f[i] = 0.f;
    c.v = __builtin_amdgcn_wmma_f32_16x16x32_bf16(false, a.v, false, bb.v,
                                                  (short)0, c.v, false, false);
#pragma unroll
    for (int i = 0; i < 8; i++) sc[i + 8 * half][nt * 16 + r] = c.f[i] * scale;
  }
  // mask padded key columns (200..223) -> softmax weight exactly 0
  for (int idx = lane; idx < 16 * 24; idx += 32) sc[idx / 24][200 + idx % 24] = -1e30f;
  __syncthreads();

  // softmax: lane owns row r, half of the 224 columns; pair-reduce via shfl
  {
    float mx = -1e30f;
    for (int j = 0; j < 112; j++) mx = fmaxf(mx, sc[r][half * 112 + j]);
    mx = fmaxf(mx, __shfl_xor(mx, 16, 32));
    float sum = 0.f;
    for (int j = 0; j < 112; j++) {
      int cc = half * 112 + j;
      float e = __expf(sc[r][cc] - mx);
      sc[r][cc] = e; sum += e;
    }
    sum += __shfl_xor(sum, 16, 32);
    float inv = 1.f / sum;
    for (int j = 0; j < 112; j++) sc[r][half * 112 + j] *= inv;
  }
  __syncthreads();

  // out[16 x 32] = P[16 x 224] * V[224 x 32], 7 K-steps, 2 N-tiles
  FragC o0, o1;
#pragma unroll
  for (int i = 0; i < 8; i++) { o0.f[i] = 0.f; o1.f[i] = 0.f; }
  long long vbase = (long long)b * D * SP + (long long)(hh * 32) * SP;
  for (int ks = 0; ks < 7; ks++) {
    FragA aa;
    const float* s0 = &sc[r][ks * 32 + 16 * half];
    const float* s1 = &sc[r][ks * 32 + 16 * half + 8];
    float4 f0 = *(const float4*)(s0);
    float4 f1 = *(const float4*)(s0 + 4);
    float4 f2 = *(const float4*)(s1);
    float4 f3 = *(const float4*)(s1 + 4);
    aa.s[0]  = f2bf(f0.x); aa.s[1]  = f2bf(f0.y); aa.s[2]  = f2bf(f0.z); aa.s[3]  = f2bf(f0.w);
    aa.s[4]  = f2bf(f1.x); aa.s[5]  = f2bf(f1.y); aa.s[6]  = f2bf(f1.z); aa.s[7]  = f2bf(f1.w);
    aa.s[8]  = f2bf(f2.x); aa.s[9]  = f2bf(f2.y); aa.s[10] = f2bf(f2.z); aa.s[11] = f2bf(f2.w);
    aa.s[12] = f2bf(f3.x); aa.s[13] = f2bf(f3.y); aa.s[14] = f2bf(f3.z); aa.s[15] = f2bf(f3.w);

    const unsigned short* v0 = vT + vbase + (long long)(r) * SP + ks * 32 + 16 * half;
    const unsigned short* v1 = vT + vbase + (long long)(16 + r) * SP + ks * 32 + 16 * half;
    FragA b0, b1;
    b0.u[0] = ((const uint4*)v0)[0]; b0.u[1] = ((const uint4*)v0)[1];
    b1.u[0] = ((const uint4*)v1)[0]; b1.u[1] = ((const uint4*)v1)[1];
    o0.v = __builtin_amdgcn_wmma_f32_16x16x32_bf16(false, aa.v, false, b0.v, (short)0, o0.v, false, false);
    o1.v = __builtin_amdgcn_wmma_f32_16x16x32_bf16(false, aa.v, false, b1.v, (short)0, o1.v, false, false);
  }
#pragma unroll
  for (int i = 0; i < 8; i++) {
    int m = m0 + i + 8 * half;
    if (m < S) {
      long long orow = (long long)b * S * D + (long long)m * D + hh * 32;
      ao[orow + r]      = f2bf(o0.f[i]);
      ao[orow + 16 + r] = f2bf(o1.f[i]);
    }
  }
}

// ---------------------------------------------------------------------------
// Elementwise / reduction helpers
// ---------------------------------------------------------------------------
DEV float blockReduceSum192(float v, float* red) {
  int t = threadIdx.x;
  red[t] = v; __syncthreads();
  for (int off = 128; off > 0; off >>= 1) {
    if (t < off && t + off < 192) red[t] += red[t + off];
    __syncthreads();
  }
  float s = red[0]; __syncthreads();
  return s;
}

// LN per 192-wide row with per-ROI params, then GELU + pos_emb.  grid = B*S
__global__ void __launch_bounds__(192)
ln_tok_kernel(const float* __restrict__ in, const float* __restrict__ g,
              const float* __restrict__ b, const float* __restrict__ pos,
              float* __restrict__ outF, unsigned short* __restrict__ outB)
{
  __shared__ float red[192];
  int t = threadIdx.x;
  long long row = blockIdx.x;
  int rr = (int)(row % 200);
  float x = in[row * 192 + t];
  float mean = blockReduceSum192(x, red) * (1.f / 192.f);
  float d = x - mean;
  float var = blockReduceSum192(d * d, red) * (1.f / 192.f);
  float y = d * rsqrtf(var + 1e-5f) * g[rr * 192 + t] + b[rr * 192 + t];
  y = gelu_exact(y) + pos[rr * 192 + t];
  outF[row * 192 + t] = y;
  outB[row * 192 + t] = f2bf(y);
}

__global__ void __launch_bounds__(192)
ln_plain_kernel(const float* __restrict__ in, const float* __restrict__ g,
                const float* __restrict__ b,
                float* __restrict__ outF, unsigned short* __restrict__ outB)
{
  __shared__ float red[192];
  int t = threadIdx.x;
  long long row = blockIdx.x;
  float x = in[row * 192 + t];
  float mean = blockReduceSum192(x, red) * (1.f / 192.f);
  float d = x - mean;
  float var = blockReduceSum192(d * d, red) * (1.f / 192.f);
  float y = d * rsqrtf(var + 1e-5f) * g[t] + b[t];
  outF[row * 192 + t] = y;
  outB[row * 192 + t] = f2bf(y);
}

// mean over S=200 tokens. grid = B, block = 192
__global__ void __launch_bounds__(192)
meanpool_kernel(const float* __restrict__ h, unsigned short* __restrict__ qm)
{
  int t = threadIdx.x, b = blockIdx.x;
  float s = 0.f;
  for (int ss = 0; ss < 200; ss++) s += h[((long long)b * 200 + ss) * 192 + t];
  qm[b * 192 + t] = f2bf(s * (1.f / 200.f));
}

// pooling attention (Q=1 per head). grid = (H, B), block = 64
__global__ void __launch_bounds__(64)
pool_attn_kernel(const float* __restrict__ qp, const unsigned short* __restrict__ kp,
                 const unsigned short* __restrict__ vp, unsigned short* __restrict__ pooled)
{
  __shared__ float red[64];
  __shared__ float accs[64][32];
  int t = threadIdx.x, hh = blockIdx.x, b = blockIdx.y;
  float qv[32];
#pragma unroll
  for (int d = 0; d < 32; d++) qv[d] = qp[b * 192 + hh * 32 + d];
  float sl[4];
  float mx = -1e30f;
  int cnt = 0;
  for (int s = t; s < 200; s += 64) {
    const unsigned short* krow = kp + ((long long)b * 200 + s) * 192 + hh * 32;
    float dot = 0.f;
#pragma unroll
    for (int d = 0; d < 32; d++) dot += qv[d] * bf2f(krow[d]);
    dot *= 0.17677669529663687f;
    sl[cnt++] = dot;
    mx = fmaxf(mx, dot);
  }
  red[t] = mx; __syncthreads();
  for (int off = 32; off > 0; off >>= 1) { if (t < off) red[t] = fmaxf(red[t], red[t + off]); __syncthreads(); }
  mx = red[0]; __syncthreads();
  float sum = 0.f; cnt = 0;
  for (int s = t; s < 200; s += 64) { float e = __expf(sl[cnt] - mx); sl[cnt] = e; sum += e; cnt++; }
  red[t] = sum; __syncthreads();
  for (int off = 32; off > 0; off >>= 1) { if (t < off) red[t] += red[t + off]; __syncthreads(); }
  float inv = 1.f / red[0]; __syncthreads();
  float acc[32];
#pragma unroll
  for (int d = 0; d < 32; d++) acc[d] = 0.f;
  cnt = 0;
  for (int s = t; s < 200; s += 64) {
    float p = sl[cnt++] * inv;
    const unsigned short* vrow = vp + ((long long)b * 200 + s) * 192 + hh * 32;
#pragma unroll
    for (int d = 0; d < 32; d++) acc[d] += p * bf2f(vrow[d]);
  }
#pragma unroll
  for (int d = 0; d < 32; d++) accs[t][d] = acc[d];
  __syncthreads();
  for (int off = 32; off > 0; off >>= 1) {
    if (t < off) { for (int d = 0; d < 32; d++) accs[t][d] += accs[t + off][d]; }
    __syncthreads();
  }
  if (t < 32) pooled[b * 192 + hh * 32 + t] = f2bf(accs[0][t]);
}

// classifier head: LN + 192->96->48->2 with GELU. grid = B, block = 192
__global__ void __launch_bounds__(192)
classifier_kernel(const float* __restrict__ p2, const float* __restrict__ g,
                  const float* __restrict__ bl,
                  const float* __restrict__ W1, const float* __restrict__ b1,
                  const float* __restrict__ W2, const float* __restrict__ b2,
                  const float* __restrict__ W3, const float* __restrict__ b3,
                  float* __restrict__ out)
{
  __shared__ float red[192];
  __shared__ float z[192]; __shared__ float z1[96]; __shared__ float z2[48];
  int t = threadIdx.x, b = blockIdx.x;
  float x = p2[b * 192 + t];
  float mean = blockReduceSum192(x, red) * (1.f / 192.f);
  float d = x - mean;
  float var = blockReduceSum192(d * d, red) * (1.f / 192.f);
  z[t] = d * rsqrtf(var + 1e-5f) * g[t] + bl[t];
  __syncthreads();
  if (t < 96) { float a = b1[t]; for (int k2 = 0; k2 < 192; k2++) a += z[k2] * W1[k2 * 96 + t]; z1[t] = gelu_exact(a); }
  __syncthreads();
  if (t < 48) { float a = b2[t]; for (int k2 = 0; k2 < 96; k2++) a += z1[k2] * W2[k2 * 48 + t]; z2[t] = gelu_exact(a); }
  __syncthreads();
  if (t < 2)  { float a = b3[t]; for (int k2 = 0; k2 < 48; k2++) a += z2[k2] * W3[k2 * 2 + t]; out[b * 2 + t] = a; }
}

// ---------------------------------------------------------------------------
// Data-prep kernels
// ---------------------------------------------------------------------------
// W[K,N] f32 -> W^T[N,Kp] bf16 (zero-padded K). grid = (ceil(N*Kp/256), batch)
__global__ void wt_transpose_kernel(const float* __restrict__ src,
                                    unsigned short* __restrict__ dst,
                                    int K, int N, int Kp)
{
  long long total = (long long)N * Kp;
  long long idx = (long long)blockIdx.x * blockDim.x + threadIdx.x;
  if (idx >= total) return;
  int n = (int)(idx / Kp), kk = (int)(idx % Kp);
  const float* s = src + (long long)blockIdx.y * K * N;
  unsigned short* d = dst + (long long)blockIdx.y * total;
  float v = (kk < K) ? s[(long long)kk * N + n] : 0.f;
  d[idx] = f2bf(v);
}

// x[256, 200*199] -> A_roi[r][b][224] bf16 (K padded 199->224)
__global__ void build_aroi_kernel(const float* __restrict__ x, unsigned short* __restrict__ dst)
{
  long long idx = (long long)blockIdx.x * 256 + threadIdx.x;
  const long long total = 200LL * 256 * 224;
  if (idx >= total) return;
  int rr = (int)(idx / (256 * 224));
  int rem = (int)(idx % (256 * 224));
  int bb = rem / 224, kk = rem % 224;
  float v = (kk < 199) ? x[(long long)bb * 39800 + rr * 199 + kk] : 0.f;
  dst[idx] = f2bf(v);
}

// zero the S-padding (s in [200,224)) of V^T[b][192][224]
__global__ void fill_vpad_kernel(unsigned short* __restrict__ vT)
{
  long long idx = (long long)blockIdx.x * 256 + threadIdx.x;
  const long long total = 256LL * 192 * 24;
  if (idx >= total) return;
  int bb = (int)(idx / (192 * 24));
  int rem = (int)(idx % (192 * 24));
  int dd = rem / 24, sp = rem % 24;
  vT[(long long)bb * 192 * 224 + (long long)dd * 224 + 200 + sp] = 0;
}

// ---------------------------------------------------------------------------
// Host orchestration
// ---------------------------------------------------------------------------
extern "C" void kernel_launch(void* const* d_in, const int* in_sizes, int n_in,
                              void* d_out, int out_size, void* d_ws, size_t ws_size,
                              hipStream_t stream)
{
  (void)in_sizes; (void)n_in; (void)out_size; (void)ws_size;
  const float* x         = (const float*)d_in[0];
  const float* roi_W     = (const float*)d_in[1];
  const float* roi_b     = (const float*)d_in[2];
  const float* roi_ln_g  = (const float*)d_in[3];
  const float* roi_ln_b  = (const float*)d_in[4];
  const float* pos_emb   = (const float*)d_in[5];
  const float* enc_Wqkv  = (const float*)d_in[6];
  const float* enc_bqkv  = (const float*)d_in[7];
  const float* enc_Wo    = (const float*)d_in[8];
  const float* enc_bo    = (const float*)d_in[9];
  const float* enc_ln1_g = (const float*)d_in[10];
  const float* enc_ln1_b = (const float*)d_in[11];
  const float* enc_W1    = (const float*)d_in[12];
  const float* enc_b1    = (const float*)d_in[13];
  const float* enc_W2    = (const float*)d_in[14];
  const float* enc_b2    = (const float*)d_in[15];
  const float* enc_ln2_g = (const float*)d_in[16];
  const float* enc_ln2_b = (const float*)d_in[17];
  const float* pool_Wqkv = (const float*)d_in[18];
  const float* pool_bqkv = (const float*)d_in[19];
  const float* pool_Wo   = (const float*)d_in[20];
  const float* pool_bo   = (const float*)d_in[21];
  const float* cls_ln_g  = (const float*)d_in[22];
  const float* cls_ln_b  = (const float*)d_in[23];
  const float* cls_W1    = (const float*)d_in[24];
  const float* cls_b1    = (const float*)d_in[25];
  const float* cls_W2    = (const float*)d_in[26];
  const float* cls_b2    = (const float*)d_in[27];
  const float* cls_W3    = (const float*)d_in[28];
  const float* cls_b3    = (const float*)d_in[29];
  float* out = (float*)d_out;

  const long long BS = 256LL * 200;  // 51200 rows

  char* base = (char*)d_ws;
  size_t off = 0;
  auto alloc = [&](size_t bytes) -> char* {
    char* p = base + off;
    off = (off + bytes + 255) & ~(size_t)255;
    return p;
  };
  unsigned short* roiWT = (unsigned short*)alloc(200ULL * 192 * 224 * 2);
  unsigned short* Aroi  = (unsigned short*)alloc(200ULL * 256 * 224 * 2);
  unsigned short* qkvT  = (unsigned short*)alloc(9ULL * 192 * 192 * 2);
  unsigned short* woT   = (unsigned short*)alloc(3ULL * 192 * 192 * 2);
  unsigned short* w1T   = (unsigned short*)alloc(3ULL * 576 * 192 * 2);
  unsigned short* w2T   = (unsigned short*)alloc(3ULL * 192 * 576 * 2);
  unsigned short* pqkvT = (unsigned short*)alloc(3ULL * 192 * 192 * 2);
  unsigned short* pwoT  = (unsigned short*)alloc(1ULL * 192 * 192 * 2);
  float* t1             = (float*)alloc(BS * 192 * 4);
  float* hF             = (float*)alloc(BS * 192 * 4);
  unsigned short* hB    = (unsigned short*)alloc(BS * 192 * 2);
  // lifetime union: {q,k,vT,ao} (attention) aliases {f1} (FFN hidden)
  char* uni             = alloc(81002496ULL);
  unsigned short* qB  = (unsigned short*)uni;
  unsigned short* kB  = (unsigned short*)(uni + 19660800);
  unsigned short* vT  = (unsigned short*)(uni + 39321600);
  unsigned short* aoB = (unsigned short*)(uni + 61341696);
  unsigned short* f1B = (unsigned short*)uni;
  unsigned short* kpB = (unsigned short*)alloc(BS * 192 * 2);
  unsigned short* vpB = (unsigned short*)alloc(BS * 192 * 2);
  unsigned short* qmB = (unsigned short*)alloc(256ULL * 192 * 2);
  float* qpF          = (float*)alloc(256ULL * 192 * 4);
  unsigned short* plB = (unsigned short*)alloc(256ULL * 192 * 2);
  float* p2F          = (float*)alloc(256ULL * 192 * 4);

  // ---- weight prep (bf16, transposed) ----
  wt_transpose_kernel<<<dim3((192 * 224 + 255) / 256, 200), 256, 0, stream>>>(roi_W, roiWT, 199, 192, 224);
  wt_transpose_kernel<<<dim3((192 * 192 + 255) / 256, 9), 256, 0, stream>>>(enc_Wqkv, qkvT, 192, 192, 192);
  wt_transpose_kernel<<<dim3((192 * 192 + 255) / 256, 3), 256, 0, stream>>>(enc_Wo, woT, 192, 192, 192);
  wt_transpose_kernel<<<dim3((576 * 192 + 255) / 256, 3), 256, 0, stream>>>(enc_W1, w1T, 192, 576, 192);
  wt_transpose_kernel<<<dim3((192 * 576 + 255) / 256, 3), 256, 0, stream>>>(enc_W2, w2T, 576, 192, 576);
  wt_transpose_kernel<<<dim3((192 * 192 + 255) / 256, 3), 256, 0, stream>>>(pool_Wqkv, pqkvT, 192, 192, 192);
  wt_transpose_kernel<<<dim3((192 * 192 + 255) / 256, 1), 256, 0, stream>>>(pool_Wo, pwoT, 192, 192, 192);
  build_aroi_kernel<<<dim3((int)((200LL * 256 * 224 + 255) / 256)), 256, 0, stream>>>(x, Aroi);

  // ---- ROI tokenizer: 200 distinct [256,199]x[199,192] GEMMs -> t1[b][r][d]
  {
    int tiles = (256 / 32) * (192 / 64);   // 24 macro-tiles
    gemm_bf16_wmma<true, false, false, true, false, false>
        <<<dim3((tiles + 3) / 4, 200), 128, 0, stream>>>(
        Aroi, 256LL * 224, 224, roiWT, 192LL * 224, 224,
        roi_b, 192, nullptr, 0, 0,
        t1, 192, 200LL * 192, nullptr, 0, 0,
        256, 192, 224, 0, 0, 0);
  }
  ln_tok_kernel<<<dim3((unsigned)BS), 192, 0, stream>>>(t1, roi_ln_g, roi_ln_b, pos_emb, hF, hB);

  int tilesBig = (int)((BS / 32) * (192 / 64));   // 4800 macro-tiles
  int tilesFF  = (int)((BS / 32) * (576 / 64));   // 14400 macro-tiles

  for (int l = 0; l < 3; l++) {
    const unsigned short* WqT = qkvT + (long long)(l * 3 + 0) * 192 * 192;
    const unsigned short* WkT = qkvT + (long long)(l * 3 + 1) * 192 * 192;
    const unsigned short* WvT = qkvT + (long long)(l * 3 + 2) * 192 * 192;
    const float* bq = enc_bqkv + (l * 3 + 0) * 192;
    const float* bk = enc_bqkv + (l * 3 + 1) * 192;
    const float* bv = enc_bqkv + (l * 3 + 2) * 192;

    gemm_bf16_wmma<true, false, false, false, true, false>
        <<<dim3((tilesBig + 3) / 4, 1), 128, 0, stream>>>(
        hB, 0, 192, WqT, 0, 192, bq, 0, nullptr, 0, 0,
        nullptr, 0, 0, qB, 0, 192, (int)BS, 192, 192, 0, 0, 0);
    gemm_bf16_wmma<true, false, false, false, true, false>
        <<<dim3((tilesBig + 3) / 4, 1), 128, 0, stream>>>(
        hB, 0, 192, WkT, 0, 192, bk, 0, nullptr, 0, 0,
        nullptr, 0, 0, kB, 0, 192, (int)BS, 192, 192, 0, 0, 0);
    fill_vpad_kernel<<<dim3((256 * 192 * 24 + 255) / 256), 256, 0, stream>>>(vT);
    gemm_bf16_wmma<true, false, false, false, true, true>     // V stored transposed [b][d][s]
        <<<dim3((tilesBig + 3) / 4, 1), 128, 0, stream>>>(
        hB, 0, 192, WvT, 0, 192, bv, 0, nullptr, 0, 0,
        nullptr, 0, 0, vT, 0, 0, (int)BS, 192, 192, 200, 224, 192LL * 224);

    attn_fused<<<dim3(13, 6, 256), 32, 0, stream>>>(qB, kB, vT, aoB);

    gemm_bf16_wmma<true, true, false, true, false, false>     // Wo + bias + residual
        <<<dim3((tilesBig + 3) / 4, 1), 128, 0, stream>>>(
        aoB, 0, 192, woT + (long long)l * 192 * 192, 0, 192, enc_bo + l * 192, 0,
        hF, 0, 192, t1, 0, 192, nullptr, 0, 0, (int)BS, 192, 192, 0, 0, 0);
    ln_plain_kernel<<<dim3((unsigned)BS), 192, 0, stream>>>(t1, enc_ln1_g + l * 192, enc_ln1_b + l * 192, hF, hB);

    gemm_bf16_wmma<true, false, true, false, true, false>     // FF1 + GELU -> bf16
        <<<dim3((tilesFF + 3) / 4, 1), 128, 0, stream>>>(
        hB, 0, 192, w1T + (long long)l * 576 * 192, 0, 192, enc_b1 + l * 576, 0,
        nullptr, 0, 0, nullptr, 0, 0, f1B, 0, 576, (int)BS, 576, 192, 0, 0, 0);
    gemm_bf16_wmma<true, true, false, true, false, false>     // FF2 + bias + residual
        <<<dim3((tilesBig + 3) / 4, 1), 128, 0, stream>>>(
        f1B, 0, 576, w2T + (long long)l * 192 * 576, 0, 576, enc_b2 + l * 192, 0,
        hF, 0, 192, t1, 0, 192, nullptr, 0, 0, (int)BS, 192, 576, 0, 0, 0);
    ln_plain_kernel<<<dim3((unsigned)BS), 192, 0, stream>>>(t1, enc_ln2_g + l * 192, enc_ln2_b + l * 192, hF, hB);
  }

  // ---- attention pooling ----
  meanpool_kernel<<<dim3(256), 192, 0, stream>>>(hF, qmB);
  {
    int tiles = (256 / 32) * (192 / 64);
    gemm_bf16_wmma<true, false, false, true, false, false>    // q projection of mean
        <<<dim3((tiles + 3) / 4, 1), 128, 0, stream>>>(
        qmB, 0, 192, pqkvT, 0, 192, pool_bqkv, 0, nullptr, 0, 0,
        qpF, 0, 192, nullptr, 0, 0, 256, 192, 192, 0, 0, 0);
  }
  gemm_bf16_wmma<true, false, false, false, true, false>      // pool K
      <<<dim3((tilesBig + 3) / 4, 1), 128, 0, stream>>>(
      hB, 0, 192, pqkvT + 36864, 0, 192, pool_bqkv + 192, 0, nullptr, 0, 0,
      nullptr, 0, 0, kpB, 0, 192, (int)BS, 192, 192, 0, 0, 0);
  gemm_bf16_wmma<true, false, false, false, true, false>      // pool V
      <<<dim3((tilesBig + 3) / 4, 1), 128, 0, stream>>>(
      hB, 0, 192, pqkvT + 2 * 36864, 0, 192, pool_bqkv + 384, 0, nullptr, 0, 0,
      nullptr, 0, 0, vpB, 0, 192, (int)BS, 192, 192, 0, 0, 0);
  pool_attn_kernel<<<dim3(6, 256), 64, 0, stream>>>(qpF, kpB, vpB, plB);
  {
    int tiles = (256 / 32) * (192 / 64);
    gemm_bf16_wmma<true, false, false, true, false, false>    // pool Wo
        <<<dim3((tiles + 3) / 4, 1), 128, 0, stream>>>(
        plB, 0, 192, pwoT, 0, 192, pool_bo, 0, nullptr, 0, 0,
        p2F, 0, 192, nullptr, 0, 0, 256, 192, 192, 0, 0, 0);
  }

  // ---- classifier ----
  classifier_kernel<<<dim3(256), 192, 0, stream>>>(p2F, cls_ln_g, cls_ln_b,
                                                   cls_W1, cls_b1, cls_W2, cls_b2,
                                                   cls_W3, cls_b3, out);
}
